// NPR_37847251812709
// MI455X (gfx1250) — matrix operand order repaired and checked by reading, not compiled, not run on hardware
//
#include <hip/hip_runtime.h>

#define EPSV    1e-6f
#define BN_EPSV 1e-5f

constexpr int B_ = 16, L_ = 2048, C_ = 256, K_ = 64, I_ = 4096;

typedef __attribute__((ext_vector_type(2)))  float  v2f;
typedef __attribute__((ext_vector_type(8)))  float  v8f;
typedef __attribute__((ext_vector_type(16))) __bf16 v16bf;

union BFV { unsigned int u[8]; v16bf v; };

__device__ __forceinline__ unsigned short f2bf(float f) {
  unsigned int u = __float_as_uint(f);
  u += 0x7FFFu + ((u >> 16) & 1u);   // round-to-nearest-even
  return (unsigned short)(u >> 16);
}

__device__ __forceinline__ v8f wmma_f32(v2f a, v2f b, v8f c) {
  return __builtin_amdgcn_wmma_f32_16x16x4_f32(false, a, false, b, (short)0, c, false, false);
}
__device__ __forceinline__ v8f wmma_bf16(v16bf a, v16bf b, v8f c) {
  return __builtin_amdgcn_wmma_f32_16x16x32_bf16(false, a, false, b, (short)0, c, false, false);
}

// ---------------------------------------------------------------- init / zero
__global__ __launch_bounds__(256) void k_init(const float* __restrict__ mu0, float* __restrict__ mu,
                                              float* __restrict__ colsum, float* __restrict__ sumsq) {
  int idx = blockIdx.x * 256 + threadIdx.x;          // over B*K*C
  mu[idx] = mu0[idx & (K_ * C_ - 1)];
  if (idx < B_ * K_) { colsum[idx] = 0.f; sumsq[idx] = 0.f; }
}

__global__ __launch_bounds__(256) void k_zero(float* __restrict__ p) {
  p[blockIdx.x * 256 + threadIdx.x] = 0.f;
}

// --------------------------------------------- logits = x . mu^T ; softmax(K)
// one block = one batch x 64-row L tile; f32 WMMA, C reduced in 64-wide chunks
__global__ __launch_bounds__(256) void k_logits(const float* __restrict__ x, const float* __restrict__ mu,
                                                float* __restrict__ softz, float* __restrict__ colsum) {
  constexpr int XS = 68, SS = 68;
  __shared__ float xs[64 * XS];
  __shared__ float ms[64 * XS];
  __shared__ float S [64 * SS];
  const int tid  = threadIdx.x;
  const int b    = blockIdx.x >> 5;                  // 32 L-tiles per batch
  const int l0   = (blockIdx.x & 31) << 6;
  const int wave = tid >> 5, lane = tid & 31, hl = lane >> 4, ln = lane & 15;

  v8f zero = {0,0,0,0,0,0,0,0};
  v8f acc[2] = {zero, zero};

  for (int cc = 0; cc < C_; cc += 64) {
    for (int idx = tid; idx < 64 * 64; idx += 256) {
      int c = idx & 63, r = idx >> 6;
      xs[r * XS + c] = x [(size_t)(b * L_ + l0 + r) * C_ + cc + c];
      ms[r * XS + c] = mu[(size_t)(b * K_ + r)      * C_ + cc + c];
    }
    __syncthreads();
#pragma unroll
    for (int t2 = 0; t2 < 2; ++t2) {
      const int tile = wave + 8 * t2, tm = tile >> 2, tn = tile & 3;
      const float* ar = &xs[(tm * 16 + ln) * XS + 2 * hl];
      const float* br = &ms[(tn * 16 + ln) * XS + 2 * hl];   // B[kc][n] = mu[n][kc]
#pragma unroll 4
      for (int kk = 0; kk < 64; kk += 4) {
        v2f a;  a.x  = ar[kk]; a.y  = ar[kk + 1];
        v2f bb; bb.x = br[kk]; bb.y = br[kk + 1];
        acc[t2] = wmma_f32(a, bb, acc[t2]);
      }
    }
    __syncthreads();
  }
#pragma unroll
  for (int t2 = 0; t2 < 2; ++t2) {
    const int tile = wave + 8 * t2, tm = tile >> 2, tn = tile & 3;
#pragma unroll
    for (int r = 0; r < 8; ++r)
      S[(tm * 16 + r + 8 * hl) * SS + tn * 16 + ln] = acc[t2][r];
  }
  __syncthreads();
  if (tid < 64) {                                    // row softmax over K=64
    float mx = -3.0e38f;
    for (int k = 0; k < 64; ++k) mx = fmaxf(mx, S[tid * SS + k]);
    float sum = 0.f;
    for (int k = 0; k < 64; ++k) { float e = expf(S[tid * SS + k] - mx); S[tid * SS + k] = e; sum += e; }
    float inv = 1.f / sum;
    for (int k = 0; k < 64; ++k) S[tid * SS + k] *= inv;
  }
  __syncthreads();
  if (tid < 64) {                                    // per-column partial sums over L
    float cs = 0.f;
    for (int r = 0; r < 64; ++r) cs += S[r * SS + tid];
    atomicAdd(&colsum[b * K_ + tid], cs);
  }
  for (int idx = tid; idx < 64 * 64; idx += 256) {
    int k = idx & 63, r = idx >> 6;
    softz[(size_t)(b * L_ + l0 + r) * K_ + k] = S[r * SS + k];
  }
}

// ----------------------- mu_unnorm[k,c] = (softz^T . x)/(EPS+colsum) ; sumsq
__global__ __launch_bounds__(256) void k_muupd(const float* __restrict__ x, const float* __restrict__ softz,
                                               const float* __restrict__ colsum, float* __restrict__ mu,
                                               float* __restrict__ sumsq) {
  constexpr int AS = 36, BS = 68, TS = 68;
  __shared__ float As[64 * AS];                      // A[k][l] chunk (32 l)
  __shared__ float Bs[32 * BS];                      // B[l][c] chunk
  __shared__ float Ts[64 * TS];
  const int tid  = threadIdx.x;
  const int b    = blockIdx.x >> 2, c0 = (blockIdx.x & 3) << 6;
  const int wave = tid >> 5, lane = tid & 31, hl = lane >> 4, ln = lane & 15;
  v8f zero = {0,0,0,0,0,0,0,0};
  v8f acc[2] = {zero, zero};

  for (int l0 = 0; l0 < L_; l0 += 32) {
    for (int idx = tid; idx < 64 * 32; idx += 256) {
      int k = idx & 63, l = idx >> 6;
      As[k * AS + l] = softz[(size_t)(b * L_ + l0 + l) * K_ + k];
    }
    for (int idx = tid; idx < 32 * 64; idx += 256) {
      int c = idx & 63, l = idx >> 6;
      Bs[l * BS + c] = x[(size_t)(b * L_ + l0 + l) * C_ + c0 + c];
    }
    __syncthreads();
#pragma unroll
    for (int t2 = 0; t2 < 2; ++t2) {
      const int tile = wave + 8 * t2, tm = tile >> 2, tn = tile & 3;
      const float* ar = &As[(tm * 16 + ln) * AS + 2 * hl];
#pragma unroll
      for (int kk = 0; kk < 32; kk += 4) {
        v2f a;  a.x  = ar[kk]; a.y = ar[kk + 1];
        v2f bb; bb.x = Bs[(kk + 2 * hl) * BS + tn * 16 + ln];
        bb.y = Bs[(kk + 2 * hl + 1) * BS + tn * 16 + ln];
        acc[t2] = wmma_f32(a, bb, acc[t2]);
      }
    }
    __syncthreads();
  }
#pragma unroll
  for (int t2 = 0; t2 < 2; ++t2) {
    const int tile = wave + 8 * t2, tm = tile >> 2, tn = tile & 3;
#pragma unroll
    for (int r = 0; r < 8; ++r) {
      int kr = tm * 16 + r + 8 * hl;
      int cc = tn * 16 + ln;
      float t = acc[t2][r] / (EPSV + colsum[b * K_ + kr]);
      mu[(size_t)(b * K_ + kr) * C_ + c0 + cc] = t;
      Ts[kr * TS + cc] = t;
    }
  }
  __syncthreads();
  if (tid < 64) {
    float s = 0.f;
    for (int c = 0; c < 64; ++c) { float v = Ts[tid * TS + c]; s += v * v; }
    atomicAdd(&sumsq[b * K_ + tid], s);
  }
}

// --------------------------------------------------- mu /= (EPS + ||mu||_2)
__global__ __launch_bounds__(256) void k_munorm(float* __restrict__ mu, float* __restrict__ sumsq,
                                                float* __restrict__ colsum, float* __restrict__ mu_out,
                                                int do_zero, int write_out) {
  const int bk  = blockIdx.x;                        // one (b,k) row per block (C==256)
  const int idx = bk * 256 + threadIdx.x;
  float n = sqrtf(sumsq[bk]);
  float v = mu[idx] / (EPSV + n);
  mu[idx] = v;
  if (write_out) mu_out[idx] = v;
  __syncthreads();
  if (do_zero && threadIdx.x == 0) { colsum[bk] = 0.f; sumsq[bk] = 0.f; }
}

// ------------------------------------------------------- recon_pre = z . mu
__global__ __launch_bounds__(256) void k_recon(const float* __restrict__ softz, const float* __restrict__ colsum,
                                               const float* __restrict__ mu, float* __restrict__ recon) {
  constexpr int ZS = 68, MS = 68;
  __shared__ float Zs[64 * ZS];
  __shared__ float Ms[64 * MS];
  const int tid  = threadIdx.x;
  const int b    = blockIdx.x >> 7;
  const int rem  = blockIdx.x & 127;
  const int l0   = (rem >> 2) << 6, c0 = (rem & 3) << 6;
  const int wave = tid >> 5, lane = tid & 31, hl = lane >> 4, ln = lane & 15;
  for (int idx = tid; idx < 64 * 64; idx += 256) {
    int k = idx & 63, l = idx >> 6;
    Zs[l * ZS + k] = softz[(size_t)(b * L_ + l0 + l) * K_ + k] / (EPSV + colsum[b * K_ + k]);
  }
  for (int idx = tid; idx < 64 * 64; idx += 256) {
    int c = idx & 63, k = idx >> 6;
    Ms[k * MS + c] = mu[(size_t)(b * K_ + k) * C_ + c0 + c];
  }
  __syncthreads();
  v8f zero = {0,0,0,0,0,0,0,0};
  v8f acc[2] = {zero, zero};
#pragma unroll
  for (int t2 = 0; t2 < 2; ++t2) {
    const int tile = wave + 8 * t2, tm = tile >> 2, tn = tile & 3;
    const float* ar = &Zs[(tm * 16 + ln) * ZS + 2 * hl];
#pragma unroll 4
    for (int kk = 0; kk < 64; kk += 4) {
      v2f a;  a.x  = ar[kk]; a.y = ar[kk + 1];
      v2f bb; bb.x = Ms[(kk + 2 * hl) * MS + tn * 16 + ln];
      bb.y = Ms[(kk + 2 * hl + 1) * MS + tn * 16 + ln];
      acc[t2] = wmma_f32(a, bb, acc[t2]);
    }
  }
#pragma unroll
  for (int t2 = 0; t2 < 2; ++t2) {
    const int tile = wave + 8 * t2, tm = tile >> 2, tn = tile & 3;
#pragma unroll
    for (int r = 0; r < 8; ++r)
      recon[(size_t)(b * L_ + l0 + tm * 16 + r + 8 * hl) * C_ + c0 + tn * 16 + ln] = acc[t2][r];
  }
}

// ------------------------------------------------ attn[b,0,k,l] (transposed)
__global__ __launch_bounds__(256) void k_attn(const float* __restrict__ softz, const float* __restrict__ colsum,
                                              float* __restrict__ attn) {
  int idx = blockIdx.x * 256 + threadIdx.x;          // over B*K*L
  int b = idx >> 17;                                 // K*L = 2^17
  int k = (idx >> 11) & 63;
  int l = idx & 2047;
  attn[idx] = softz[(size_t)(b * L_ + l) * K_ + k] / (EPSV + colsum[b * K_ + k]);
}

// --------------------------- fused MLP (bf16 WMMA, f32 accum), in-place tile
__global__ __launch_bounds__(256) void k_mlp(float* __restrict__ recon,
    const float* __restrict__ w1, const float* __restrict__ b1,
    const float* __restrict__ gam, const float* __restrict__ bet,
    const float* __restrict__ bnm, const float* __restrict__ bnv,
    const float* __restrict__ w2, const float* __restrict__ b2) {
  constexpr int XS = 260, WS1 = 264, WS2 = 40, HS = 40;
  __shared__ __align__(4) unsigned short xb[64 * XS];    // x tile, bf16, resident
  __shared__ __align__(4) unsigned short wbuf[256 * WS2];// time-shared: w1 [32][WS1] or w2T [256][WS2]
  __shared__ __align__(4) unsigned short hds[64 * HS];   // h chunk, bf16
  __shared__ float sc[32];
  __shared__ float sh[32];
  const int tid  = threadIdx.x;
  const size_t t0 = (size_t)blockIdx.x * 64;             // token row base (B*L rows total)
  const int wave = tid >> 5, lane = tid & 31, hl = lane >> 4, ln = lane & 15;

  for (int idx = tid; idx < 64 * 256; idx += 256) {      // stage whole tile before any write
    int r = idx >> 8, c = idx & 255;
    xb[r * XS + c] = f2bf(recon[(t0 + r) * C_ + c]);
  }
  v8f zero = {0,0,0,0,0,0,0,0};
  v8f acc[8];
#pragma unroll
  for (int j = 0; j < 8; ++j) acc[j] = zero;

  for (int ic = 0; ic < I_; ic += 32) {
    for (int idx = tid; idx < 32 * 256; idx += 256) {    // w1 chunk [i][c]
      int i = idx >> 8, c = idx & 255;
      wbuf[i * WS1 + c] = f2bf(w1[(size_t)(ic + i) * C_ + c]);
    }
    if (tid < 32) {
      int i = tid;
      float s = gam[ic + i] * rsqrtf(bnv[ic + i] + BN_EPSV);
      sc[i] = s;
      sh[i] = bet[ic + i] + (b1[ic + i] - bnm[ic + i]) * s;
    }
    __syncthreads();
    {   // GEMM1: h[64,32] = x . w1c^T ; 8 tiles (4x2), one per wave
      const int hm = wave >> 1, hn = wave & 1;
      v8f hacc = zero;
      const unsigned short* ar = &xb[(hm * 16 + ln) * XS];
      const unsigned short* br = &wbuf[(hn * 16 + ln) * WS1 + 16 * hl];  // B[kc][n]=w1c[n][kc]
#pragma unroll
      for (int kk = 0; kk < 256; kk += 32) {
        BFV a, bb;
#pragma unroll
        for (int v = 0; v < 8; ++v) {
          const int ko = 2 * v + 16 * hl;
          a.u[v]  = *(const unsigned int*)&ar[kk + ko];
          bb.u[v] = *(const unsigned int*)&br[kk + 2 * v];
        }
        hacc = wmma_bf16(a.v, bb.v, hacc);
      }
#pragma unroll
      for (int r = 0; r < 8; ++r) {                      // BN + ReLU epilogue
        int tr = hm * 16 + r + 8 * hl;
        int ii = hn * 16 + ln;
        float hv = fmaxf(hacc[r] * sc[ii] + sh[ii], 0.f);
        hds[tr * HS + ii] = f2bf(hv);
      }
    }
    __syncthreads();
    for (int idx = tid; idx < 256 * 32; idx += 256) {    // w2 chunk transposed -> [c][i]
      int i = idx & 31, c = idx >> 5;
      wbuf[c * WS2 + i] = f2bf(w2[(size_t)c * I_ + ic + i]);
    }
    __syncthreads();
#pragma unroll
    for (int j = 0; j < 8; ++j) {                        // GEMM2: out += h . w2c^T (K=32)
      const int tt = wave + 8 * j, tm = tt >> 4, tn = tt & 15;
      const unsigned short* ar = &hds[(tm * 16 + ln) * HS];
      const unsigned short* br = &wbuf[(tn * 16 + ln) * WS2 + 16 * hl];
      BFV a, bb;
#pragma unroll
      for (int v = 0; v < 8; ++v) {
        const int ko = 2 * v + 16 * hl;
        a.u[v]  = *(const unsigned int*)&ar[ko];
        bb.u[v] = *(const unsigned int*)&br[2 * v];
      }
      acc[j] = wmma_bf16(a.v, bb.v, acc[j]);
    }
    __syncthreads();
  }
#pragma unroll
  for (int j = 0; j < 8; ++j) {
    const int tt = wave + 8 * j, tm = tt >> 4, tn = tt & 15;
#pragma unroll
    for (int r = 0; r < 8; ++r) {
      size_t row = t0 + tm * 16 + r + 8 * hl;
      int col = tn * 16 + ln;
      recon[row * C_ + col] = acc[j][r] + b2[col];
    }
  }
}

// ---------------------------------------------------------------------- host
extern "C" void kernel_launch(void* const* d_in, const int* in_sizes, int n_in,
                              void* d_out, int out_size, void* d_ws, size_t ws_size,
                              hipStream_t stream) {
  (void)in_sizes; (void)n_in; (void)out_size; (void)ws_size;
  const float* x   = (const float*)d_in[0];
  const float* mu0 = (const float*)d_in[2];
  const float* w1  = (const float*)d_in[3];
  const float* b1  = (const float*)d_in[4];
  const float* gam = (const float*)d_in[5];
  const float* bet = (const float*)d_in[6];
  const float* bnm = (const float*)d_in[7];
  const float* bnv = (const float*)d_in[8];
  const float* w2  = (const float*)d_in[9];
  const float* b2  = (const float*)d_in[10];

  float* mu_out   = (float*)d_out;
  float* attn_out = mu_out + (size_t)B_ * K_ * C_;
  float* recon    = attn_out + (size_t)B_ * K_ * L_;

  float* wsf    = (float*)d_ws;
  float* mu_ws  = wsf;                                   //  1.0 MB
  float* softz  = mu_ws  + (size_t)B_ * K_ * C_;         //  8.0 MB
  float* colsum = softz  + (size_t)B_ * L_ * K_;         //  4 KB
  float* sumsq  = colsum + B_ * K_;                      //  4 KB

  k_init<<<(B_ * K_ * C_) / 256, 256, 0, stream>>>(mu0, mu_ws, colsum, sumsq);
  for (int s = 0; s < 10; ++s) {
    k_logits<<<B_ * (L_ / 64), 256, 0, stream>>>(x, mu_ws, softz, colsum);
    k_muupd <<<B_ * (C_ / 64), 256, 0, stream>>>(x, softz, colsum, mu_ws, sumsq);
    k_munorm<<<B_ * K_,        256, 0, stream>>>(mu_ws, sumsq, colsum, mu_out,
                                                 (s < 9) ? 1 : 0, (s == 9) ? 1 : 0);
  }
  // recon_pre = z10 . mu_final (needs surviving stage-10 softz/colsum), written into d_out slice
  k_recon<<<B_ * (L_ / 64) * (C_ / 64), 256, 0, stream>>>(softz, colsum, mu_ws, recon);
  // final attention map with mu_final
  k_zero  <<<(B_ * K_) / 256, 256, 0, stream>>>(colsum);
  k_logits<<<B_ * (L_ / 64), 256, 0, stream>>>(x, mu_ws, softz, colsum);
  k_attn  <<<(B_ * K_ * L_) / 256, 256, 0, stream>>>(softz, colsum, attn_out);
  // fused MLP in place on the recon slice
  k_mlp<<<(B_ * L_) / 64, 256, 0, stream>>>(recon, w1, b1, gam, bet, bnm, bnv, w2, b2);
}
